// MoEFFN_56925496541862
// MI455X (gfx1250) — compile-verified, matrix operand order checked by reading, not run on hardware
//
#include <hip/hip_runtime.h>
#include <hip/hip_bf16.h>
#include <math.h>

typedef __attribute__((ext_vector_type(16))) __bf16 v16bf;
typedef __attribute__((ext_vector_type(8)))  float  v8f;

#define D_DIM 1024
#define H_DIM 4096
#define E_NUM 8
#define N_TOK 4096            // B*S = 2*2048
#define ROWS_TOTAL (N_TOK*2)  // top-2 routing -> 8192 compacted rows

#define MT 128
#define NT 128
#define KT 32
#define LDK 36                // padded LDS K-stride (bf16): 72B rows -> conflict-free, 8B aligned

// ---------------------------------------------------------------- init
__global__ void moe_init_kernel(float* out, int n, int* cnts) {
    int i = blockIdx.x * blockDim.x + threadIdx.x;
    if (i < n) out[i] = 0.0f;
    if (i < 16) cnts[i] = 0;  // counts[8] + cursors[8]
}

// ---------------------------------------------------------------- router
__global__ __launch_bounds__(256)
void moe_router_kernel(const float* __restrict__ x, const float* __restrict__ rw,
                       int* counts, int* top_e, float* top_w) {
    int wave = threadIdx.x >> 5;
    int lane = threadIdx.x & 31;
    int t = blockIdx.x * 8 + wave;
    if (t >= N_TOK) return;
    const float* xp = x + (size_t)t * D_DIM;
    float acc[E_NUM];
#pragma unroll
    for (int e = 0; e < E_NUM; ++e) acc[e] = 0.0f;
    for (int d = lane; d < D_DIM; d += 32) {
        float xv = xp[d];
        const float4* r4 = (const float4*)(rw + (size_t)d * E_NUM);
        float4 a = r4[0], b = r4[1];
        acc[0] += xv * a.x; acc[1] += xv * a.y; acc[2] += xv * a.z; acc[3] += xv * a.w;
        acc[4] += xv * b.x; acc[5] += xv * b.y; acc[6] += xv * b.z; acc[7] += xv * b.w;
    }
#pragma unroll
    for (int e = 0; e < E_NUM; ++e)
#pragma unroll
        for (int off = 16; off > 0; off >>= 1)
            acc[e] += __shfl_xor(acc[e], off, 32);
    if (lane == 0) {
        float m = acc[0];
#pragma unroll
        for (int e = 1; e < E_NUM; ++e) m = fmaxf(m, acc[e]);
        float p[E_NUM], s = 0.0f;
#pragma unroll
        for (int e = 0; e < E_NUM; ++e) { p[e] = __expf(acc[e] - m); s += p[e]; }
        float inv_s = 1.0f / s;
#pragma unroll
        for (int e = 0; e < E_NUM; ++e) p[e] *= inv_s;
        int e0 = 0;
#pragma unroll
        for (int e = 1; e < E_NUM; ++e) if (p[e] > p[e0]) e0 = e;
        int e1 = (e0 == 0) ? 1 : 0;
#pragma unroll
        for (int e = 0; e < E_NUM; ++e) if (e != e0 && p[e] > p[e1]) e1 = e;
        float w0 = p[e0], w1 = p[e1];
        float inv = 1.0f / (w0 + w1);
        top_e[2 * t + 0] = e0;  top_e[2 * t + 1] = e1;
        top_w[2 * t + 0] = w0 * inv;  top_w[2 * t + 1] = w1 * inv;
        atomicAdd(&counts[e0], 1);
        atomicAdd(&counts[e1], 1);
    }
}

// ---------------------------------------------------------------- scan
__global__ void moe_scan_kernel(const int* counts, int* offsets) {
    if (threadIdx.x == 0 && blockIdx.x == 0) {
        int s = 0;
        for (int e = 0; e < E_NUM; ++e) { offsets[e] = s; s += counts[e]; }
    }
}

// ---------------------------------------------------------------- place
__global__ void moe_place_kernel(const int* top_e, const float* top_w, const int* offsets,
                                 int* cursors, int* rowmap, float* roww) {
    int t = blockIdx.x * blockDim.x + threadIdx.x;
    if (t >= N_TOK) return;
#pragma unroll
    for (int k = 0; k < 2; ++k) {
        int e = top_e[2 * t + k];
        int pos = offsets[e] + atomicAdd(&cursors[e], 1);
        rowmap[pos] = t;
        roww[pos] = top_w[2 * t + k];
    }
}

// ------------------------------------------------- WMMA fragment loader
// 16-bit A/B frag (ISA 7.12.2): lane (l&15)=row/col, (l>>4) selects K-halves;
// VGPR q holds K pair k0 = 2(q&3) + 16(q>>2) + 8*(l>>4)
__device__ inline v16bf load_frag(const __bf16* p, int hl) {
    union { v16bf v; unsigned u[8]; } f;
#pragma unroll
    for (int q = 0; q < 8; ++q) {
        int k0 = ((q & 3) * 2) + ((q >> 2) * 16) + hl * 8;
        f.u[q] = *(const unsigned*)(p + k0);
    }
    return f.v;
}

// pack 4 fp32 -> 4 bf16 -> one 8B LDS store
__device__ inline void pack_store4(__bf16* dst, float a, float b, float c, float d) {
    union { __bf16 h[4]; unsigned long long u; } pk;
    pk.h[0] = (__bf16)a; pk.h[1] = (__bf16)b; pk.h[2] = (__bf16)c; pk.h[3] = (__bf16)d;
    *(unsigned long long*)dst = pk.u;
}

// ---------------------------------------------------------------- GEMM1: h = gelu(x_gathered @ w1[e] + b1[e])
__global__ __launch_bounds__(256)
void moe_gemm1_kernel(const float* __restrict__ x, const float* __restrict__ w1,
                      const float* __restrict__ b1,
                      const int* counts, const int* offsets, const int* rowmap,
                      __bf16* __restrict__ hbuf) {
    __shared__ __align__(16) __bf16 As[2][MT * LDK];
    __shared__ __align__(16) __bf16 Bs[2][NT * LDK];
    int e = blockIdx.z;
    int cnt = counts[e];
    int mt = blockIdx.y;
    if (mt * MT >= cnt) return;
    int base = offsets[e];
    int n0 = blockIdx.x * NT;
    const float* W = w1 + (size_t)e * D_DIM * H_DIM;
    const float* b1e = b1 + (size_t)e * H_DIM;

    int tid = threadIdx.x;
    // A loader: thread -> (row ar, 16-wide k-run at ah)
    int ar = tid >> 1, ah = (tid & 1) * 16;
    int lrA = mt * MT + ar;
    int tok = (lrA < cnt) ? rowmap[base + lrA] : 0;
    const float* arow = x + (size_t)tok * D_DIM + ah;
    // B loader: thread -> (col nb, 16-long k-run at bh); coalesced over nb
    int nb = tid & 127, bh = (tid >> 7) * 16;

    int wv = tid >> 5, lane = tid & 31;
    int wr = wv >> 2, wc = wv & 3;
    int r = lane & 15, hl = lane >> 4;

    v8f acc[4][2];
#pragma unroll
    for (int i = 0; i < 4; ++i)
#pragma unroll
        for (int j = 0; j < 2; ++j)
            acc[i][j] = (v8f){0.f, 0.f, 0.f, 0.f, 0.f, 0.f, 0.f, 0.f};

    float areg[16], breg[16];
    // ---- prologue: stage tile 0 into buffer 0
#pragma unroll
    for (int j = 0; j < 16; ++j) areg[j] = arow[j];
#pragma unroll
    for (int j = 0; j < 16; ++j) breg[j] = W[(size_t)(bh + j) * H_DIM + n0 + nb];
#pragma unroll
    for (int j = 0; j < 16; j += 4)
        pack_store4(&As[0][ar * LDK + ah + j], areg[j], areg[j+1], areg[j+2], areg[j+3]);
#pragma unroll
    for (int j = 0; j < 16; j += 4)
        pack_store4(&Bs[0][nb * LDK + bh + j], breg[j], breg[j+1], breg[j+2], breg[j+3]);

    const int NKT = D_DIM / KT;
    for (int kt = 0; kt < NKT; ++kt) {
        __syncthreads();
        int cur = kt & 1;
        bool more = (kt + 1) < NKT;
        if (more) {  // prefetch next tile into registers (latency hidden by WMMAs)
            int k = (kt + 1) * KT;
#pragma unroll
            for (int j = 0; j < 16; ++j) areg[j] = arow[k + j];
#pragma unroll
            for (int j = 0; j < 16; ++j) breg[j] = W[(size_t)(k + bh + j) * H_DIM + n0 + nb];
        }
        v16bf afrag[4], bfrag[2];
#pragma unroll
        for (int i = 0; i < 4; ++i)
            afrag[i] = load_frag(&As[cur][((wr * 4 + i) * 16 + r) * LDK], hl);
#pragma unroll
        for (int j = 0; j < 2; ++j)
            bfrag[j] = load_frag(&Bs[cur][((wc * 2 + j) * 16 + r) * LDK], hl);
#pragma unroll
        for (int i = 0; i < 4; ++i)
#pragma unroll
            for (int j = 0; j < 2; ++j)
                acc[i][j] = __builtin_amdgcn_wmma_f32_16x16x32_bf16(
                    false, afrag[i], false, bfrag[j], (short)0, acc[i][j], false, false);
        if (more) {
#pragma unroll
            for (int j = 0; j < 16; j += 4)
                pack_store4(&As[cur ^ 1][ar * LDK + ah + j], areg[j], areg[j+1], areg[j+2], areg[j+3]);
#pragma unroll
            for (int j = 0; j < 16; j += 4)
                pack_store4(&Bs[cur ^ 1][nb * LDK + bh + j], breg[j], breg[j+1], breg[j+2], breg[j+3]);
        }
    }

    // bias + exact GELU (erf), store bf16 h
#pragma unroll
    for (int j = 0; j < 2; ++j) {
        int n = n0 + (wc * 2 + j) * 16 + r;
        float bias = b1e[n];
#pragma unroll
        for (int i = 0; i < 4; ++i) {
#pragma unroll
            for (int v = 0; v < 8; ++v) {
                int lr = mt * MT + (wr * 4 + i) * 16 + hl * 8 + v;
                if (lr < cnt) {
                    float hv = acc[i][j][v] + bias;
                    hv = 0.5f * hv * (1.0f + erff(hv * 0.70710678118f));
                    hbuf[(size_t)(base + lr) * H_DIM + n] = (__bf16)hv;
                }
            }
        }
    }
}

// ---------------------------------------------------------------- GEMM2: out[tok] += w * (h @ w2[e] + b2[e])
__global__ __launch_bounds__(256)
void moe_gemm2_kernel(const __bf16* __restrict__ hbuf, const float* __restrict__ w2,
                      const float* __restrict__ b2,
                      const int* counts, const int* offsets, const int* rowmap,
                      const float* roww, float* __restrict__ out) {
    __shared__ __align__(16) __bf16 As[2][MT * LDK];
    __shared__ __align__(16) __bf16 Bs[2][NT * LDK];
    int e = blockIdx.z;
    int cnt = counts[e];
    int mt = blockIdx.y;
    if (mt * MT >= cnt) return;
    int base = offsets[e];
    int n0 = blockIdx.x * NT;
    const float* W = w2 + (size_t)e * H_DIM * D_DIM;
    const float* b2e = b2 + (size_t)e * D_DIM;

    int tid = threadIdx.x;
    int ar = tid >> 1, ah = (tid & 1) * 16;
    int nb = tid & 127, bh = (tid >> 7) * 16;
    int wv = tid >> 5, lane = tid & 31;
    int wr = wv >> 2, wc = wv & 3;
    int r = lane & 15, hl = lane >> 4;

    // hbuf has (ROWS_TOTAL + MT) rows of capacity, so tile overrun reads are safe
    const __bf16* arow = hbuf + (size_t)(base + mt * MT + ar) * H_DIM + ah;

    v8f acc[4][2];
#pragma unroll
    for (int i = 0; i < 4; ++i)
#pragma unroll
        for (int j = 0; j < 2; ++j)
            acc[i][j] = (v8f){0.f, 0.f, 0.f, 0.f, 0.f, 0.f, 0.f, 0.f};

    unsigned long long areg[4];
    float breg[16];
    // ---- prologue: stage tile 0 into buffer 0
#pragma unroll
    for (int q = 0; q < 4; ++q) areg[q] = *(const unsigned long long*)(arow + q * 4);
#pragma unroll
    for (int j = 0; j < 16; ++j) breg[j] = W[(size_t)(bh + j) * D_DIM + n0 + nb];
#pragma unroll
    for (int q = 0; q < 4; ++q)
        *(unsigned long long*)&As[0][ar * LDK + ah + q * 4] = areg[q];
#pragma unroll
    for (int j = 0; j < 16; j += 4)
        pack_store4(&Bs[0][nb * LDK + bh + j], breg[j], breg[j+1], breg[j+2], breg[j+3]);

    const int NKT = H_DIM / KT;
    for (int kt = 0; kt < NKT; ++kt) {
        __syncthreads();
        int cur = kt & 1;
        bool more = (kt + 1) < NKT;
        if (more) {
            int k = (kt + 1) * KT;
#pragma unroll
            for (int q = 0; q < 4; ++q) areg[q] = *(const unsigned long long*)(arow + k + q * 4);
#pragma unroll
            for (int j = 0; j < 16; ++j) breg[j] = W[(size_t)(k + bh + j) * D_DIM + n0 + nb];
        }
        v16bf afrag[4], bfrag[2];
#pragma unroll
        for (int i = 0; i < 4; ++i)
            afrag[i] = load_frag(&As[cur][((wr * 4 + i) * 16 + r) * LDK], hl);
#pragma unroll
        for (int j = 0; j < 2; ++j)
            bfrag[j] = load_frag(&Bs[cur][((wc * 2 + j) * 16 + r) * LDK], hl);
#pragma unroll
        for (int i = 0; i < 4; ++i)
#pragma unroll
            for (int j = 0; j < 2; ++j)
                acc[i][j] = __builtin_amdgcn_wmma_f32_16x16x32_bf16(
                    false, afrag[i], false, bfrag[j], (short)0, acc[i][j], false, false);
        if (more) {
#pragma unroll
            for (int q = 0; q < 4; ++q)
                *(unsigned long long*)&As[cur ^ 1][ar * LDK + ah + q * 4] = areg[q];
#pragma unroll
            for (int j = 0; j < 16; j += 4)
                pack_store4(&Bs[cur ^ 1][nb * LDK + bh + j], breg[j], breg[j+1], breg[j+2], breg[j+3]);
        }
    }

    // weighted combine into out via fp32 global atomics
#pragma unroll
    for (int i = 0; i < 4; ++i) {
#pragma unroll
        for (int v = 0; v < 8; ++v) {
            int lr = mt * MT + (wr * 4 + i) * 16 + hl * 8 + v;
            if (lr < cnt) {
                int grow = base + lr;
                float wgt = roww[grow];
                int tok = rowmap[grow];
#pragma unroll
                for (int j = 0; j < 2; ++j) {
                    int n = n0 + (wc * 2 + j) * 16 + r;
                    float val = (acc[i][j][v] + b2e[n]) * wgt;
                    atomicAdd(&out[(size_t)tok * D_DIM + n], val);
                }
            }
        }
    }
}

// ---------------------------------------------------------------- launch
extern "C" void kernel_launch(void* const* d_in, const int* in_sizes, int n_in,
                              void* d_out, int out_size, void* d_ws, size_t ws_size,
                              hipStream_t stream) {
    const float* x        = (const float*)d_in[0];
    const float* router_w = (const float*)d_in[1];
    const float* w1       = (const float*)d_in[2];
    const float* b1       = (const float*)d_in[3];
    const float* w2       = (const float*)d_in[4];
    const float* b2       = (const float*)d_in[5];
    float* out = (float*)d_out;

    char* ws = (char*)d_ws;
    int*   counts  = (int*)(ws + 0);
    int*   cursors = (int*)(ws + 32);
    int*   offsets = (int*)(ws + 64);
    int*   top_e   = (int*)(ws + 128);
    float* top_w   = (float*)(ws + 128 + (size_t)2 * N_TOK * 4);
    int*   rowmap  = (int*)(ws + 128 + (size_t)4 * N_TOK * 4);
    float* roww    = (float*)(ws + 128 + (size_t)4 * N_TOK * 4 + (size_t)ROWS_TOTAL * 4);
    size_t hoff = 128 + (size_t)4 * N_TOK * 4 + (size_t)ROWS_TOTAL * 8;
    hoff = (hoff + 255) & ~(size_t)255;
    __bf16* hbuf = (__bf16*)(ws + hoff);  // (ROWS_TOTAL + MT) x H_DIM bf16 ~ 65 MB

    moe_init_kernel<<<(out_size + 255) / 256, 256, 0, stream>>>(out, out_size, counts);
    moe_router_kernel<<<N_TOK / 8, 256, 0, stream>>>(x, router_w, counts, top_e, top_w);
    moe_scan_kernel<<<1, 32, 0, stream>>>(counts, offsets);
    moe_place_kernel<<<N_TOK / 256, 256, 0, stream>>>(top_e, top_w, offsets, cursors, rowmap, roww);

    dim3 g1(H_DIM / NT, N_TOK / MT, E_NUM);
    moe_gemm1_kernel<<<g1, 256, 0, stream>>>(x, w1, b1, counts, offsets, rowmap, hbuf);
    dim3 g2(D_DIM / NT, N_TOK / MT, E_NUM);
    moe_gemm2_kernel<<<g2, 256, 0, stream>>>(hbuf, w2, b2, counts, offsets, rowmap, roww, out);
}
